// MultiHeadAttention_82935818486382
// MI455X (gfx1250) — matrix operand order repaired and checked by reading, not compiled, not run on hardware
//
#include <hip/hip_runtime.h>
#include <hip/hip_bf16.h>

typedef __attribute__((ext_vector_type(16))) _Float16 v16h;
typedef __attribute__((ext_vector_type(8)))  _Float16 v8h;
typedef __attribute__((ext_vector_type(4)))  _Float16 v4h;
typedef __attribute__((ext_vector_type(8)))  float    v8f;
typedef __attribute__((ext_vector_type(4)))  float    v4f;
typedef unsigned int u32;
typedef unsigned int u32x4 __attribute__((ext_vector_type(4)));
typedef int          i32x8 __attribute__((ext_vector_type(8)));
typedef int          i32x4 __attribute__((ext_vector_type(4)));

#define B_   4
#define S_   2048
#define D_   1024
#define H_   16
#define HD_  64
#define BS_  (B_ * S_)      // 8192 rows
#define D3_  (3 * D_)       // 3072

#if __has_builtin(__builtin_amdgcn_tensor_load_to_lds) && \
    __has_builtin(__builtin_amdgcn_s_wait_tensorcnt)
#define HAVE_TDM 1
#else
#define HAVE_TDM 0
#endif

// D = A(16x32 f16) * B(32x16 f16) + C(16x16 f32)
__device__ __forceinline__ v8f wmma_f32_f16(v16h a, v16h b, v8f c) {
  return __builtin_amdgcn_wmma_f32_16x16x32_f16(
      false, a, false, b, (short)0, c, false, false);
}

// A-fragment (16x32): lane row = lane&15; e<8 -> K=8*hi+e ; e>=8 -> K=16+8*hi+e-8
__device__ __forceinline__ v16h frag_a(const _Float16* base, int ld) {
  const int lane = threadIdx.x & 31;
  const int row  = lane & 15;
  const int hi   = lane >> 4;
  const _Float16* p = base + row * ld + hi * 8;
  v8h lo = *(const v8h*)(p);
  v8h hh = *(const v8h*)(p + 16);
  return __builtin_shufflevector(lo, hh, 0,1,2,3,4,5,6,7,8,9,10,11,12,13,14,15);
}

// B-fragment (32x16): lane col = lane&15; e<8 -> K=8*hi+e ; e>=8 -> K=16+8*hi+e-8
// Memory: base[n*ld + k] == B[k][n]   (works for global or LDS pointers)
__device__ __forceinline__ v16h frag_b(const _Float16* base, int ld) {
  const int lane = threadIdx.x & 31;
  const int col  = lane & 15;
  const int hi   = lane >> 4;
  const _Float16* p = base + col * ld + hi * 8;
  v8h lo = *(const v8h*)(p);
  v8h hh = *(const v8h*)(p + 16);
  return __builtin_shufflevector(lo, hh, 0,1,2,3,4,5,6,7,8,9,10,11,12,13,14,15);
}

#if HAVE_TDM
// Issue a 2-D TDM tile load: tile_h rows x tile_w f16 elements, row stride =
// tensor_w elements, from global gbase into LDS byte offset lds_off.
// D# packing per CDNA5 ISA 8.3/8.4 (data_size=1 -> 2 bytes, type=2, count=1).
__device__ __forceinline__ void tdm_load_tile_f16(
    const _Float16* gbase, u32 lds_off,
    u32 tensor_w, u32 tensor_h, u32 tile_w, u32 tile_h) {
  unsigned long long ga = (unsigned long long)(size_t)gbase;
  u32x4 g0;
  g0.x = 1u;                                   // count=1, user descriptor
  g0.y = lds_off;                              // lds_addr (bytes)
  g0.z = (u32)ga;                              // global_addr[31:0]
  g0.w = (u32)(ga >> 32) | (2u << 30);         // global_addr[56:32] | type=2
  i32x8 g1;
  g1[0] = (int)(1u << 16);                     // data_size=1 (2 bytes)
  g1[1] = (int)((tensor_w & 0xffffu) << 16);   // tensor_dim0[15:0]
  g1[2] = (int)((tensor_w >> 16) | ((tensor_h & 0xffffu) << 16)); // dim0 hi | dim1 lo
  g1[3] = (int)((tensor_h >> 16) | (tile_w << 16));               // dim1 hi | tile_dim0
  g1[4] = (int)(tile_h & 0xffffu);             // tile_dim1 (tile_dim2 = 0)
  g1[5] = (int)tensor_w;                       // tensor_dim0_stride[31:0]
  g1[6] = 0;                                   // stride hi | dim1_stride lo
  g1[7] = 0;
  i32x4 z4 = {0, 0, 0, 0};
#if defined(__clang_major__) && (__clang_major__ >= 23)
  i32x8 z8 = {0, 0, 0, 0, 0, 0, 0, 0};
  __builtin_amdgcn_tensor_load_to_lds(g0, g1, z4, z4, z8, 0);
#else
  __builtin_amdgcn_tensor_load_to_lds(g0, g1, z4, z4, 0);
#endif
}
#endif

// Fallback cooperative staging (used when TDM builtin unavailable)
__device__ __forceinline__ void stage64x64_b256(const _Float16* __restrict__ g, int gld,
                                                _Float16* __restrict__ l) {
  const int t    = threadIdx.x;        // 256 threads, 64x64 tile
  const int row  = t >> 2;
  const int colb = (t & 3) * 16;
#pragma unroll
  for (int c = 0; c < 2; ++c)
    *(v8h*)(l + row * 64 + colb + c * 8) =
        *(const v8h*)(g + (size_t)row * gld + colb + c * 8);
}
__device__ __forceinline__ void stage64x64_b128(const _Float16* __restrict__ g, int gld,
                                                _Float16* __restrict__ l) {
  const int t    = threadIdx.x & 127;  // 128 threads, 64x64 tile
  const int row  = t >> 1;
  const int colb = (t & 1) * 32;
#pragma unroll
  for (int c = 0; c < 4; ++c)
    *(v8h*)(l + row * 64 + colb + c * 8) =
        *(const v8h*)(g + (size_t)row * gld + colb + c * 8);
}

// ---------------------------------------------------------------------------
// Stage 0: fp32 -> f16 convert, 4-wide
// ---------------------------------------------------------------------------
__global__ void cvt_f32_to_f16_v4(const float* __restrict__ in,
                                  _Float16* __restrict__ out, int n4) {
  int i = blockIdx.x * blockDim.x + threadIdx.x;
  int stride = gridDim.x * blockDim.x;
  for (; i < n4; i += stride) {
    v4f v = ((const v4f*)in)[i];
    ((v4h*)out)[i] = __builtin_convertvector(v, v4h);
  }
}

// ---------------------------------------------------------------------------
// GEMM tiles: B tile 64(n) x 64(k) f16 = 8 KiB, double buffered via TDM.
// Block 256 thr = 8 waves, tile 256(m) x 64(n); wave -> 32x64, 16 wmma/k64-step.
// ---------------------------------------------------------------------------
#define GEMM_TILE_ELEMS (64 * 64)
#define GEMM_TILE_BYTES (GEMM_TILE_ELEMS * 2)    // 8 KiB

__global__ void __launch_bounds__(256)
mha_qkv_gemm(const _Float16* __restrict__ Xh, const _Float16* __restrict__ Wq,
             const float* __restrict__ bias,
             _Float16* __restrict__ Qh, _Float16* __restrict__ Kh,
             _Float16* __restrict__ Vt) {
  extern __shared__ _Float16 smem_q[];   // 2 x (64x64) f16 = 16 KiB
  const int lane = threadIdx.x & 31;
  const int wave = threadIdx.x >> 5;
  const int hi   = lane >> 4;
  const int col  = lane & 15;
  const int n0   = blockIdx.x * 64;
  const int m0   = blockIdx.y * 256 + wave * 32;

  const _Float16* pa = Xh + (size_t)m0 * D_;
  const _Float16* pw = Wq + (size_t)n0 * D_;

  v8f acc[8] = {};
  v16h a00 = frag_a(pa, D_);
  v16h a01 = frag_a(pa + 32, D_);
  v16h a10 = frag_a(pa + (size_t)16 * D_, D_);
  v16h a11 = frag_a(pa + (size_t)16 * D_ + 32, D_);

#if HAVE_TDM
  if (wave == 0) tdm_load_tile_f16(pw, 0, D_, D3_, 64, 64);
#else
  stage64x64_b256(pw, D_, smem_q);
#endif

  for (int kt = 0; kt < D_; kt += 64) {
    const int cur = (kt >> 6) & 1;
    const int nxt = cur ^ 1;
    const int kn  = (kt + 64) & (D_ - 1);  // next k (wraps; last reissue unused)
#if HAVE_TDM
    if (wave == 0) {
      tdm_load_tile_f16(pw + kn, (u32)(nxt * GEMM_TILE_BYTES), D_, D3_, 64, 64);
      __builtin_amdgcn_s_wait_tensorcnt((short)1);
    }
#else
    stage64x64_b256(pw + kn, D_, smem_q + nxt * GEMM_TILE_ELEMS);
#endif
    __syncthreads();                       // buffer `cur` ready for all waves

    __builtin_prefetch(pa + (size_t)(lane & 15) * D_ + ((kt + 256) & (D_ - 1)), 0, 3);
    v16h an00 = frag_a(pa + kn, D_);
    v16h an01 = frag_a(pa + kn + 32, D_);
    v16h an10 = frag_a(pa + (size_t)16 * D_ + kn, D_);
    v16h an11 = frag_a(pa + (size_t)16 * D_ + kn + 32, D_);

    const _Float16* bb = smem_q + cur * GEMM_TILE_ELEMS;
    {   // k sub-chunk 0
      v16h b0 = frag_b(bb + 0 * 16 * 64, 64);
      v16h b1 = frag_b(bb + 1 * 16 * 64, 64);
      v16h b2 = frag_b(bb + 2 * 16 * 64, 64);
      v16h b3 = frag_b(bb + 3 * 16 * 64, 64);
      acc[0] = wmma_f32_f16(a00, b0, acc[0]);
      acc[1] = wmma_f32_f16(a00, b1, acc[1]);
      acc[2] = wmma_f32_f16(a00, b2, acc[2]);
      acc[3] = wmma_f32_f16(a00, b3, acc[3]);
      acc[4] = wmma_f32_f16(a10, b0, acc[4]);
      acc[5] = wmma_f32_f16(a10, b1, acc[5]);
      acc[6] = wmma_f32_f16(a10, b2, acc[6]);
      acc[7] = wmma_f32_f16(a10, b3, acc[7]);
    }
    {   // k sub-chunk 1
      v16h b0 = frag_b(bb + 0 * 16 * 64 + 32, 64);
      v16h b1 = frag_b(bb + 1 * 16 * 64 + 32, 64);
      v16h b2 = frag_b(bb + 2 * 16 * 64 + 32, 64);
      v16h b3 = frag_b(bb + 3 * 16 * 64 + 32, 64);
      acc[0] = wmma_f32_f16(a01, b0, acc[0]);
      acc[1] = wmma_f32_f16(a01, b1, acc[1]);
      acc[2] = wmma_f32_f16(a01, b2, acc[2]);
      acc[3] = wmma_f32_f16(a01, b3, acc[3]);
      acc[4] = wmma_f32_f16(a11, b0, acc[4]);
      acc[5] = wmma_f32_f16(a11, b1, acc[5]);
      acc[6] = wmma_f32_f16(a11, b2, acc[6]);
      acc[7] = wmma_f32_f16(a11, b3, acc[7]);
    }
    a00 = an00; a01 = an01; a10 = an10; a11 = an11;
    __syncthreads();                       // all waves done with `cur`
  }

#pragma unroll
  for (int mi = 0; mi < 2; ++mi) {
#pragma unroll
    for (int j = 0; j < 4; ++j) {
      const int N    = n0 + j * 16 + col;
      const float bn = bias[N];
      const int sec  = N >> 10;            // 0=Q 1=K 2=V
      const int head = (N & (D_ - 1)) >> 6;
      const int hd   = N & (HD_ - 1);
#pragma unroll
      for (int r = 0; r < 8; ++r) {
        const int M  = m0 + mi * 16 + r + 8 * hi;
        const int bt = M >> 11;            // / S_
        const int s  = M & (S_ - 1);
        const float v = acc[mi * 4 + j][r] + bn;
        if (sec == 0) {
          Qh[(((size_t)(bt * H_ + head) * S_) + s) * HD_ + hd] = (_Float16)(v * 0.125f);
        } else if (sec == 1) {
          Kh[(((size_t)(bt * H_ + head) * S_) + s) * HD_ + hd] = (_Float16)v;
        } else {
          Vt[(((size_t)(bt * H_ + head) * HD_) + hd) * S_ + s] = (_Float16)v;
        }
      }
    }
  }
}

// ---------------------------------------------------------------------------
// Stage 2: flash attention. Grid (S/64, H, B), 128 thr = 4 waves.
// K tile (64keys x 64hd) and V^T tile (64hd x 64keys) staged via TDM into
// double-buffered LDS shared by all 4 waves; block-uniform key-step count.
// ---------------------------------------------------------------------------
#define ATT_TILE_ELEMS (64 * 64)
#define ATT_TILE_BYTES (ATT_TILE_ELEMS * 2)      // 8 KiB
#define ATT_VOFF_BYTES (2 * ATT_TILE_BYTES)      // V buffers after 2 K buffers
#define ATT_POFF_BYTES (4 * ATT_TILE_BYTES)      // P staging after K+V buffers
#define ATT_LDS_BYTES  (ATT_POFF_BYTES + 4 * 16 * 64 * 2)   // 40 KiB

__global__ void __launch_bounds__(128)
mha_flash_attn(const _Float16* __restrict__ Qh, const _Float16* __restrict__ Kh,
               const _Float16* __restrict__ Vt, const int* __restrict__ mask,
               _Float16* __restrict__ Att) {
  extern __shared__ _Float16 smem_a[];
  _Float16* Kbuf = smem_a;                              // [2][64*64]
  _Float16* Vbuf = smem_a + ATT_VOFF_BYTES / 2;         // [2][64*64]
  _Float16* Pst  = smem_a + ATT_POFF_BYTES / 2;         // [4][16*64]

  const int lane = threadIdx.x & 31;
  const int wave = threadIdx.x >> 5;
  const int hi   = lane >> 4;
  const int col  = lane & 15;
  const int qt   = blockIdx.x;
  const int h    = blockIdx.y;
  const int b    = blockIdx.z;
  const int m0   = qt * 64 + wave * 16;

  const _Float16* Qbh = Qh + (size_t)(b * H_ + h) * S_ * HD_;
  const _Float16* Kbh = Kh + (size_t)(b * H_ + h) * S_ * HD_;
  const _Float16* Vbh = Vt + (size_t)(b * H_ + h) * HD_ * S_;
  const int* mb = mask + b * S_;

  const v16h aq0 = frag_a(Qbh + (size_t)m0 * HD_ + 0, HD_);
  const v16h aq1 = frag_a(Qbh + (size_t)m0 * HD_ + 32, HD_);

  v8f o[4] = {};
  float mrow[8], lrow[8];
#pragma unroll
  for (int r = 0; r < 8; ++r) { mrow[r] = -1e30f; lrow[r] = 0.0f; }

  _Float16* pwv = Pst + wave * (16 * 64);
  const int nkb = qt + 1;                  // block-uniform 64-key steps

#if HAVE_TDM
  if (wave == 0) {
    tdm_load_tile_f16(Kbh, 0, HD_, S_, 64, 64);
    tdm_load_tile_f16(Vbh, ATT_VOFF_BYTES, S_, HD_, 64, 64);
  }
#else
  stage64x64_b128(Kbh, HD_, Kbuf);
  stage64x64_b128(Vbh, S_, Vbuf);
#endif

  for (int kt = 0; kt < nkb; ++kt) {
    const int key0 = kt * 64;
    const int cur  = kt & 1;
    const int nxt  = cur ^ 1;
    const int kyn  = ((kt + 1) < nkb ? (kt + 1) : 0) * 64;  // wrap; unused reissue
#if HAVE_TDM
    if (wave == 0) {
      tdm_load_tile_f16(Kbh + (size_t)kyn * HD_, (u32)(nxt * ATT_TILE_BYTES),
                        HD_, S_, 64, 64);
      tdm_load_tile_f16(Vbh + kyn, (u32)(ATT_VOFF_BYTES + nxt * ATT_TILE_BYTES),
                        S_, HD_, 64, 64);
      __builtin_amdgcn_s_wait_tensorcnt((short)2);
    }
#else
    stage64x64_b128(Kbh + (size_t)kyn * HD_, HD_, Kbuf + nxt * ATT_TILE_ELEMS);
    stage64x64_b128(Vbh + kyn, S_, Vbuf + nxt * ATT_TILE_ELEMS);
#endif
    __syncthreads();                        // buffers `cur` ready

    const _Float16* Kb = Kbuf + cur * ATT_TILE_ELEMS;  // [key][hd], ld 64
    const _Float16* Vb = Vbuf + cur * ATT_TILE_ELEMS;  // [hd][key], ld 64

    // S = Q K^T over 64 keys
    v8f sc[4] = {};
#pragma unroll
    for (int st = 0; st < 4; ++st) {
      v16h bk0 = frag_b(Kb + (st * 16) * 64 + 0, 64);
      v16h bk1 = frag_b(Kb + (st * 16) * 64 + 32, 64);
      sc[st] = wmma_f32_f16(aq0, bk0, sc[st]);
      sc[st] = wmma_f32_f16(aq1, bk1, sc[st]);
    }

    int keyc[4], kvalid[4];
#pragma unroll
    for (int st = 0; st < 4; ++st) {
      keyc[st]   = key0 + st * 16 + col;
      kvalid[st] = mb[keyc[st]];
    }

    float mnew[8];
#pragma unroll
    for (int r = 0; r < 8; ++r) mnew[r] = mrow[r];
#pragma unroll
    for (int st = 0; st < 4; ++st) {
#pragma unroll
      for (int r = 0; r < 8; ++r) {
        const int q = m0 + r + 8 * hi;
        const bool ok = (kvalid[st] != 0) && (keyc[st] <= q);
        float sv = ok ? sc[st][r] : -1e30f;
        sc[st][r] = sv;
        mnew[r] = fmaxf(mnew[r], sv);
      }
    }
#pragma unroll
    for (int d = 1; d < 16; d <<= 1) {
#pragma unroll
      for (int r = 0; r < 8; ++r)
        mnew[r] = fmaxf(mnew[r], __shfl_xor(mnew[r], d, 32));
    }

    float alpha[8], lsum[8];
#pragma unroll
    for (int r = 0; r < 8; ++r) {
      alpha[r] = __expf(mrow[r] - mnew[r]);
      mrow[r]  = mnew[r];
      lsum[r]  = 0.0f;
    }

    // P = exp(S-m) (masked -> 0), stage in C-layout to this wave's LDS region
#pragma unroll
    for (int st = 0; st < 4; ++st) {
#pragma unroll
      for (int r = 0; r < 8; ++r) {
        const int q = m0 + r + 8 * hi;
        const bool ok = (kvalid[st] != 0) && (keyc[st] <= q);
        float p = ok ? __expf(sc[st][r] - mnew[r]) : 0.0f;
        lsum[r] += p;
        pwv[(r + 8 * hi) * 64 + st * 16 + col] = (_Float16)p;
      }
    }
#pragma unroll
    for (int d = 1; d < 16; d <<= 1) {
#pragma unroll
      for (int r = 0; r < 8; ++r) lsum[r] += __shfl_xor(lsum[r], d, 32);
    }
#pragma unroll
    for (int r = 0; r < 8; ++r) lrow[r] = lrow[r] * alpha[r] + lsum[r];

    // wave-local LDS RAW: wait for our stores, then read P as A-fragments
    asm volatile("s_wait_dscnt 0x0" ::: "memory");

    v16h pa0, pa1;
    {
      const _Float16* p = pwv + (lane & 15) * 64 + hi * 8;
      v8h lo = *(const v8h*)(p);
      v8h hh = *(const v8h*)(p + 16);
      pa0 = __builtin_shufflevector(lo, hh, 0,1,2,3,4,5,6,7,8,9,10,11,12,13,14,15);
      v8h lo2 = *(const v8h*)(p + 32);
      v8h hh2 = *(const v8h*)(p + 48);
      pa1 = __builtin_shufflevector(lo2, hh2, 0,1,2,3,4,5,6,7,8,9,10,11,12,13,14,15);
    }

    // O = O*alpha + P V
#pragma unroll
    for (int j = 0; j < 4; ++j) {
#pragma unroll
      for (int r = 0; r < 8; ++r) o[j][r] *= alpha[r];
      v16h bv0 = frag_b(Vb + (j * 16) * 64 + 0, 64);
      o[j] = wmma_f32_f16(pa0, bv0, o[j]);
      v16h bv1 = frag_b(Vb + (j * 16) * 64 + 32, 64);
      o[j] = wmma_f32_f16(pa1, bv1, o[j]);
    }
    __syncthreads();                        // all waves done with `cur`
  }

#pragma unroll
  for (int r = 0; r < 8; ++r) {
    const int q = m0 + r + 8 * hi;
    const float inv = lrow[r] > 0.0f ? 1.0f / lrow[r] : 0.0f;
#pragma unroll
    for (int j = 0; j < 4; ++j) {
      Att[((size_t)(b * S_ + q) * D_) + h * HD_ + j * 16 + col] =
          (_Float16)(o[j][r] * inv);
    }
  }
}

// ---------------------------------------------------------------------------
// Stage 3: out projection. out[m,n] = sum_k Att[m,k]*Wout[n,k] + bias[n] (fp32)
// Same TDM-staged k64 structure as stage 1.
// ---------------------------------------------------------------------------
__global__ void __launch_bounds__(256)
mha_out_gemm(const _Float16* __restrict__ Ah, const _Float16* __restrict__ Wh,
             const float* __restrict__ bias, float* __restrict__ out) {
  extern __shared__ _Float16 smem_o[];
  const int lane = threadIdx.x & 31;
  const int wave = threadIdx.x >> 5;
  const int hi   = lane >> 4;
  const int col  = lane & 15;
  const int n0   = blockIdx.x * 64;
  const int m0   = blockIdx.y * 256 + wave * 32;

  const _Float16* pa = Ah + (size_t)m0 * D_;
  const _Float16* pw = Wh + (size_t)n0 * D_;

  v8f acc[8] = {};
  v16h a00 = frag_a(pa, D_);
  v16h a01 = frag_a(pa + 32, D_);
  v16h a10 = frag_a(pa + (size_t)16 * D_, D_);
  v16h a11 = frag_a(pa + (size_t)16 * D_ + 32, D_);

#if HAVE_TDM
  if (wave == 0) tdm_load_tile_f16(pw, 0, D_, D_, 64, 64);
#else
  stage64x64_b256(pw, D_, smem_o);
#endif

  for (int kt = 0; kt < D_; kt += 64) {
    const int cur = (kt >> 6) & 1;
    const int nxt = cur ^ 1;
    const int kn  = (kt + 64) & (D_ - 1);
#if HAVE_TDM
    if (wave == 0) {
      tdm_load_tile_f16(pw + kn, (u32)(nxt * GEMM_TILE_BYTES), D_, D_, 64, 64);
      __builtin_amdgcn_s_wait_tensorcnt((short)1);
    }
#else
    stage64x64_b256(pw + kn, D_, smem_o + nxt * GEMM_TILE_ELEMS);
#endif
    __syncthreads();

    __builtin_prefetch(pa + (size_t)(lane & 15) * D_ + ((kt + 256) & (D_ - 1)), 0, 3);
    v16h an00 = frag_a(pa + kn, D_);
    v16h an01 = frag_a(pa + kn + 32, D_);
    v16h an10 = frag_a(pa + (size_t)16 * D_ + kn, D_);
    v16h an11 = frag_a(pa + (size_t)16 * D_ + kn + 32, D_);

    const _Float16* bb = smem_o + cur * GEMM_TILE_ELEMS;
    {
      v16h b0 = frag_b(bb + 0 * 16 * 64, 64);
      v16h b1 = frag_b(bb + 1 * 16 * 64, 64);
      v16h b2 = frag_b(bb + 2 * 16 * 64, 64);
      v16h b3 = frag_b(bb + 3 * 16 * 64, 64);
      acc[0] = wmma_f32_f16(a00, b0, acc[0]);
      acc[1] = wmma_f32_f16(a00, b1, acc[1]);
      acc[2] = wmma_f32_f16(a00, b2, acc[2]);
      acc[3] = wmma_f32_f16(a00, b3, acc[3]);
      acc[4] = wmma_f32_f16(a10, b0, acc[4]);
      acc[5] = wmma_f32_f16(a10, b1, acc[5]);
      acc[6] = wmma_f32_f16(a10, b2, acc[6]);
      acc[7] = wmma_f32_f16(a10, b3, acc[7]);
    }
    {
      v16h b0 = frag_b(bb + 0 * 16 * 64 + 32, 64);
      v16h b1 = frag_b(bb + 1 * 16 * 64 + 32, 64);
      v16h b2 = frag_b(bb + 2 * 16 * 64 + 32, 64);
      v16h b3 = frag_b(bb + 3 * 16 * 64 + 32, 64);
      acc[0] = wmma_f32_f16(a01, b0, acc[0]);
      acc[1] = wmma_f32_f16(a01, b1, acc[1]);
      acc[2] = wmma_f32_f16(a01, b2, acc[2]);
      acc[3] = wmma_f32_f16(a01, b3, acc[3]);
      acc[4] = wmma_f32_f16(a11, b0, acc[4]);
      acc[5] = wmma_f32_f16(a11, b1, acc[5]);
      acc[6] = wmma_f32_f16(a11, b2, acc[6]);
      acc[7] = wmma_f32_f16(a11, b3, acc[7]);
    }
    a00 = an00; a01 = an01; a10 = an10; a11 = an11;
    __syncthreads();
  }

#pragma unroll
  for (int mi = 0; mi < 2; ++mi) {
#pragma unroll
    for (int j = 0; j < 4; ++j) {
      const int N = n0 + j * 16 + col;
      const float bn = bias[N];
#pragma unroll
      for (int r = 0; r < 8; ++r) {
        const int M = m0 + mi * 16 + r + 8 * hi;
        out[(size_t)M * D_ + N] = acc[mi * 4 + j][r] + bn;
      }
    }
  }
}

// ---------------------------------------------------------------------------
// Workspace layout (bytes)
// ---------------------------------------------------------------------------
static constexpr size_t SZ_XH   = (size_t)BS_ * D_ * 2;
static constexpr size_t SZ_WQKV = (size_t)D3_ * D_ * 2;
static constexpr size_t SZ_WOUT = (size_t)D_ * D_ * 2;
static constexpr size_t SZ_HEAD = (size_t)B_ * H_ * S_ * HD_ * 2;

static constexpr size_t OFF_XH   = 0;
static constexpr size_t OFF_WQKV = OFF_XH + SZ_XH;
static constexpr size_t OFF_WOUT = OFF_WQKV + SZ_WQKV;
static constexpr size_t OFF_Q    = OFF_WOUT + SZ_WOUT;
static constexpr size_t OFF_K    = OFF_Q + SZ_HEAD;
static constexpr size_t OFF_V    = OFF_K + SZ_HEAD;
static constexpr size_t OFF_ATT  = OFF_V + SZ_HEAD;

extern "C" void kernel_launch(void* const* d_in, const int* in_sizes, int n_in,
                              void* d_out, int out_size, void* d_ws, size_t ws_size,
                              hipStream_t stream) {
  const float* x     = (const float*)d_in[0];
  const int*   mask  = (const int*)d_in[1];
  const float* qkv_w = (const float*)d_in[2];
  const float* qkv_b = (const float*)d_in[3];
  const float* out_w = (const float*)d_in[4];
  const float* out_b = (const float*)d_in[5];
  float* out = (float*)d_out;

  char* ws = (char*)d_ws;
  _Float16* Xh   = (_Float16*)(ws + OFF_XH);
  _Float16* Wqkv = (_Float16*)(ws + OFF_WQKV);
  _Float16* Wout = (_Float16*)(ws + OFF_WOUT);
  _Float16* Qh   = (_Float16*)(ws + OFF_Q);
  _Float16* Kh   = (_Float16*)(ws + OFF_K);
  _Float16* Vt   = (_Float16*)(ws + OFF_V);
  _Float16* Att  = (_Float16*)(ws + OFF_ATT);

  {
    int n0 = (BS_ * D_) / 4;
    cvt_f32_to_f16_v4<<<(n0 + 255) / 256, 256, 0, stream>>>(x, Xh, n0);
    int n1 = (D3_ * D_) / 4;
    cvt_f32_to_f16_v4<<<(n1 + 255) / 256, 256, 0, stream>>>(qkv_w, Wqkv, n1);
    int n2 = (D_ * D_) / 4;
    cvt_f32_to_f16_v4<<<(n2 + 255) / 256, 256, 0, stream>>>(out_w, Wout, n2);
  }

  mha_qkv_gemm<<<dim3(D3_ / 64, BS_ / 256), 256, 2 * GEMM_TILE_BYTES, stream>>>(
      Xh, Wqkv, qkv_b, Qh, Kh, Vt);

  mha_flash_attn<<<dim3(S_ / 64, H_, B_), 128, ATT_LDS_BYTES, stream>>>(
      Qh, Kh, Vt, mask, Att);

  mha_out_gemm<<<dim3(D_ / 64, BS_ / 256), 256, 2 * GEMM_TILE_BYTES, stream>>>(
      Att, Wout, out_b, out);
}
